// BayesianClassifier_44624710205475
// MI455X (gfx1250) — matrix operand order repaired and checked by reading, not compiled, not run on hardware
//
#include <hip/hip_runtime.h>

typedef _Float16 v16h __attribute__((ext_vector_type(16)));
typedef _Float16 v8h  __attribute__((ext_vector_type(8)));
typedef _Float16 v4h  __attribute__((ext_vector_type(4)));
typedef float    v8f  __attribute__((ext_vector_type(8)));

#define S_SAMPLES 40
#define NB 8
#define NC 256
#define NK 14
#define HW 40000
#define TILE_PX 256
#define ROWP 264   // padded LDS row stride in halfs (256 + 8): 528B, 16B-aligned, bank-skewed

__global__ __launch_bounds__(256)
void BayesianClassifier_44624710205475_kernel(
    const float* __restrict__ feat,   // [8,256,200,200]
    const float* __restrict__ weight, // [14,256]
    const float* __restrict__ bias,   // [14]
    const int*   __restrict__ masks,  // [40,8,256]
    float* __restrict__ out)          // [8,14,200,200]
{
    __shared__ _Float16 sF[TILE_PX * ROWP];   // feature tile [pixel][c] f16  (135168 B)
    __shared__ _Float16 sW[2][16 * ROWP];     // double-buffered masked weights (16896 B)
    __shared__ _Float16 sWh[NK * NC];         // f16 weights, staged once      (7168 B)

    const int b   = blockIdx.y;
    const int p0  = blockIdx.x * TILE_PX;
    const int tid = threadIdx.x;

    // Stage weights once (f16; masked weight = f16(w) * {0,2} is exact).
    for (int i = tid; i < NK * NC; i += 256) sWh[i] = (_Float16)weight[i];

    // Stage feature tile as f16, transposed to [pixel][c].
    // lane == pixel -> coalesced global b32 loads; 4 channels packed per ds b64 store.
    const float* fb = feat + (size_t)b * NC * HW;
    {
        const int  gp = p0 + tid;
        const bool ok = gp < HW;
        _Float16* row = &sF[tid * ROWP];
        for (int c4 = 0; c4 < NC / 4; ++c4) {
            v4h h;
            #pragma unroll
            for (int j = 0; j < 4; ++j) {
                float v = ok ? fb[(size_t)(c4 * 4 + j) * HW + gp] : 0.0f;
                h[j] = (_Float16)v;
            }
            *(v4h*)(row + c4 * 4) = h;
        }
    }
    __syncthreads();

    const int lane = tid & 31;        // wave32
    const int wv   = tid >> 5;        // 0..7
    const int nn   = lane & 15;
    const int hi   = lane >> 4;       // lane group
    const int px0  = wv * 32 + nn;    // this wave's first 16-pixel column tile
    const int px1  = px0 + 16;        // second tile

    // B fragments (16-bit B 32x16): elem j -> K = ks*32 + 16*hi + j, N = px.
    // Two pixel tiles held in registers across all 40 samples.
    v16h Bf0[8], Bf1[8];
    {
        const _Float16* brow0 = &sF[px0 * ROWP];
        const _Float16* brow1 = &sF[px1 * ROWP];
        #pragma unroll
        for (int ks = 0; ks < 8; ++ks) {
            Bf0[ks] = *(const v16h*)(brow0 + ks * 32 + hi * 16);
            Bf1[ks] = *(const v16h*)(brow1 + ks * 32 + hi * 16);
        }
    }

    // D/C fragment: vgpr r -> output row k = r + 8*hi.
    float biasv[8];
    #pragma unroll
    for (int r = 0; r < 8; ++r) {
        int k = r + 8 * hi;
        biasv[r] = (k < NK) ? bias[k] : 0.0f;
    }

    v8f sum0 = {0.f,0.f,0.f,0.f,0.f,0.f,0.f,0.f};
    v8f sum1 = {0.f,0.f,0.f,0.f,0.f,0.f,0.f,0.f};

    for (int s = 0; s < S_SAMPLES; ++s) {
        _Float16* wbuf = sW[s & 1];
        // Build masked f16 weights; tid == channel (NC == blockDim.x == 256).
        {
            const int* mrow = masks + ((size_t)s * NB + b) * NC;
            _Float16 msc = mrow[tid] ? (_Float16)2.0f : (_Float16)0.0f;
            #pragma unroll
            for (int k = 0; k < 16; ++k) {
                _Float16 v = (k < NK) ? (_Float16)(sWh[k * NC + tid] * msc)
                                      : (_Float16)0.0f;
                wbuf[k * ROWP + tid] = v;
            }
        }
        __syncthreads();  // single barrier per sample (double-buffered sW)

        // GEMM: one A fragment feeds two WMMAs (two pixel tiles).
        v8f acc0 = {0.f,0.f,0.f,0.f,0.f,0.f,0.f,0.f};
        v8f acc1 = {0.f,0.f,0.f,0.f,0.f,0.f,0.f,0.f};
        const _Float16* arow = &wbuf[(lane & 15) * ROWP];  // A row m = lane&15
        #pragma unroll
        for (int ks = 0; ks < 8; ++ks) {
            int base = ks * 32;
            // 16-bit A 16x32 layout: elem j -> K = base + 16*hi + j (matches B)
            //
            v8h lo = *(const v8h*)(arow + base + 16 * hi);
            v8h hh = *(const v8h*)(arow + base + 16 * hi + 8);
            v16h Af = __builtin_shufflevector(lo, hh,
                0, 1, 2, 3, 4, 5, 6, 7, 8, 9, 10, 11, 12, 13, 14, 15);
            acc0 = __builtin_amdgcn_wmma_f32_16x16x32_f16(
                false, Af, false, Bf0[ks], (short)0, acc0, false, false);
            acc1 = __builtin_amdgcn_wmma_f32_16x16x32_f16(
                false, Af, false, Bf1[ks], (short)0, acc1, false, false);
        }

        // bias + sigmoid + accumulate (f32).
        #pragma unroll
        for (int r = 0; r < 8; ++r) {
            float x0 = acc0[r] + biasv[r];
            float x1 = acc1[r] + biasv[r];
            float e0 = __builtin_amdgcn_exp2f(-x0 * 1.44269504088896f);
            float e1 = __builtin_amdgcn_exp2f(-x1 * 1.44269504088896f);
            sum0[r] += __builtin_amdgcn_rcpf(1.0f + e0);
            sum1[r] += __builtin_amdgcn_rcpf(1.0f + e1);
        }
    }

    // Epilogue: mean, then logit(mean) = ln(m) - ln(1-m).
    #pragma unroll
    for (int t = 0; t < 2; ++t) {
        const int gp = p0 + (t == 0 ? px0 : px1);
        if (gp < HW) {
            #pragma unroll
            for (int r = 0; r < 8; ++r) {
                int k = r + 8 * hi;
                if (k < NK) {
                    float m = (t == 0 ? sum0[r] : sum1[r]) * (1.0f / S_SAMPLES);
                    m = fminf(fmaxf(m, 1e-7f), 1.0f - 1e-7f);
                    float v = 0.69314718056f *
                        (__builtin_amdgcn_logf(m) - __builtin_amdgcn_logf(1.0f - m));
                    out[((size_t)b * NK + k) * HW + gp] = v;
                }
            }
        }
    }
}

extern "C" void kernel_launch(void* const* d_in, const int* in_sizes, int n_in,
                              void* d_out, int out_size, void* d_ws, size_t ws_size,
                              hipStream_t stream) {
    (void)in_sizes; (void)n_in; (void)d_ws; (void)ws_size; (void)out_size;
    const float* feat   = (const float*)d_in[0];
    const float* weight = (const float*)d_in[1];
    const float* bias   = (const float*)d_in[2];
    const int*   masks  = (const int*)d_in[3];
    float* out = (float*)d_out;

    dim3 grid((HW + TILE_PX - 1) / TILE_PX, NB);  // 157 x 8 blocks
    dim3 block(256);                              // 8 wave32
    hipLaunchKernelGGL(BayesianClassifier_44624710205475_kernel,
                       grid, block, 0, stream, feat, weight, bias, masks, out);
}
